// RelationAwareSelfAttention_14276471292155
// MI455X (gfx1250) — matrix operand order repaired and checked by reading, not compiled, not run on hardware
//
#include <hip/hip_runtime.h>
#include <math.h>

typedef __attribute__((ext_vector_type(16))) _Float16 v16h;
typedef __attribute__((ext_vector_type(8)))  _Float16 v8h;
typedef __attribute__((ext_vector_type(8)))  float    v8f;
typedef __attribute__((ext_vector_type(4)))  float    v4f;

// Problem constants
static constexpr int CB  = 16;   // batch
static constexpr int CS  = 512;  // seq
static constexpr int CH  = 128;  // hidden
static constexpr int CNH = 4;    // heads
static constexpr int CR  = 4;    // relations
static constexpr int CHD = 32;   // head dim

// ---------------- WMMA helpers (gfx1250, wave32) ----------------

__device__ __forceinline__ v8f wmma16(v16h a, v16h b, v8f c) {
    // D = A(16x32 f16) x B(32x16 f16) + C(16x16 f32)
    return __builtin_amdgcn_wmma_f32_16x16x32_f16(false, a, false, b, (short)0, c, false, false);
}

// A fragment (16x32, f16). Per ISA: lanes 0-15 -> K {0..7,16..23}; lanes 16-31 -> K {8..15,24..31}.
__device__ __forceinline__ v16h frag_a_h(const _Float16* base, int stride, int row0, int k0, int lane) {
    const int row  = row0 + (lane & 15);
    const int koff = (lane >= 16) ? 8 : 0;
    const _Float16* p = base + (long)row * stride + k0 + koff;
    v8h lo = *(const v8h*)(p);
    v8h hi = *(const v8h*)(p + 16);
    v16h r;
#pragma unroll
    for (int i = 0; i < 8; ++i) { r[i] = lo[i]; r[8 + i] = hi[i]; }
    return r;
}

// B fragment (32x16, f16) from a [N][K] (column-major-by-N) buffer: lanes 0-15 -> K {0..7,16..23}; lanes 16-31 -> K {8..15,24..31}.
__device__ __forceinline__ v16h frag_b_h(const _Float16* base, int stride, int n0, int k0, int lane) {
    const int n    = n0 + (lane & 15);
    const int koff = (lane >= 16) ? 8 : 0;
    const _Float16* p = base + (long)n * stride + k0 + koff;
    v8h lo = *(const v8h*)(p);
    v8h hi = *(const v8h*)(p + 16);
    v16h r;
#pragma unroll
    for (int i = 0; i < 8; ++i) { r[i] = lo[i]; r[8 + i] = hi[i]; }
    return r;
}

// A fragment built from f32 data (probs in LDS), converted to f16 on the fly.
__device__ __forceinline__ v16h frag_a_f32(const float* base, int stride, int row0, int k0, int lane) {
    const int row  = row0 + (lane & 15);
    const int koff = (lane >= 16) ? 8 : 0;
    const float* p = base + (long)row * stride + k0 + koff;
    v16h r;
#pragma unroll
    for (int i = 0; i < 8; ++i) { r[i] = (_Float16)p[i]; r[8 + i] = (_Float16)p[16 + i]; }
    return r;
}

// ---------------- Tiny prep kernels ----------------

__global__ void k_wprob(const float* __restrict__ rw, float* __restrict__ wp,
                        float* __restrict__ zb) {
    const int t = threadIdx.x;
    if (t == 0) {
        float m = fmaxf(fmaxf(rw[0], rw[1]), fmaxf(rw[2], rw[3]));
        float e0 = __expf(rw[0] - m), e1 = __expf(rw[1] - m);
        float e2 = __expf(rw[2] - m), e3 = __expf(rw[3] - m);
        float s = e0 + e1 + e2 + e3;
        wp[0] = e0 / s; wp[1] = e1 / s; wp[2] = e2 / s; wp[3] = e3 / s;
    }
    for (int i = t; i < 128; i += 32) zb[i] = 0.0f;   // zero bias for m_r GEMMs
}

__global__ void k_xh(const float* __restrict__ x, _Float16* __restrict__ xh) {
    int i = blockIdx.x * 256 + threadIdx.x;
    xh[i] = (_Float16)x[i];
}

// sym[r] = E_r @ E_r  (f32, 4 x 128 x 128, trivial cost)
__global__ void k_sym(const float* __restrict__ E, float* __restrict__ sym) {
    int idx = blockIdx.x * 256 + threadIdx.x;       // 65536 total
    int r = idx >> 14;
    int i = (idx >> 7) & 127;
    int k = idx & 127;
    const float* Er = E + r * (CH * CH);
    float acc = 0.f;
#pragma unroll 8
    for (int j = 0; j < CH; ++j) acc += Er[i * CH + j] * Er[j * CH + k];
    sym[idx] = acc;
}

// ---------------- Projection GEMMs: q, k, v, m_0..m_3 ----------------
// grid (128, 1, 7), block 128 (4 waves). z: 0=q 1=k 2=v 3..6=m_r
// All destinations share the index form ((b*grp + gbase + nh)*S + s)*HD + d.

__global__ __launch_bounds__(128) void k_proj(
        const _Float16* __restrict__ xh,
        const float* __restrict__ Wq, const float* __restrict__ bq,
        const float* __restrict__ Wk, const float* __restrict__ bk,
        const float* __restrict__ Wv, const float* __restrict__ bv,
        const float* __restrict__ sym, const float* __restrict__ wp,
        const float* __restrict__ zb,
        _Float16* __restrict__ qh, _Float16* __restrict__ kh,
        _Float16* __restrict__ vh, _Float16* __restrict__ mh) {
    __shared__ _Float16 Bt[128 * 136];   // [N][K], padded rows (272B = 17*16B aligned)
    const int tid  = threadIdx.x;
    const int lane = tid & 31;
    const int wv   = tid >> 5;
    const int z    = blockIdx.z;

    // Uniform per-z configuration (no divergent branches in hot loops).
    float scale = 1.0f;
    const float* bias;
    _Float16* dst;
    int grp, gbase;
    if (z == 0)      { bias = bq; dst = qh; grp = 4;  gbase = 0;            scale = 0.17677669529663689f; }
    else if (z == 1) { bias = bk; dst = kh; grp = 4;  gbase = 0; }
    else if (z == 2) { bias = bv; dst = vh; grp = 4;  gbase = 0; }
    else             { bias = zb; dst = mh; grp = 16; gbase = (z - 3) * 4;  scale = sqrtf(wp[z - 3] * (1.0f / 32.0f)); }

    // Stage B^T into LDS: Bt[n][k].
    if (z < 3) {
        const float* W = (z == 0) ? Wq : (z == 1) ? Wk : Wv;
        for (int idx = tid; idx < 16384; idx += 128) {
            int k = idx >> 7, n = idx & 127;
            Bt[n * 136 + k] = (_Float16)W[idx];
        }
    } else {
        // m_r = x @ sym_r^T : B(K=h,N=t) = sym[t][h] -> row-major copy
        const float* Sr = sym + (z - 3) * 16384;
        for (int idx = tid; idx < 16384; idx += 128) {
            int t = idx >> 7, h = idx & 127;
            Bt[t * 136 + h] = (_Float16)Sr[idx];
        }
    }
    __syncthreads();

    const int row0 = blockIdx.x * 64 + wv * 16;   // 16 rows per wave
    v16h afr[4];
#pragma unroll
    for (int kc = 0; kc < 4; ++kc) afr[kc] = frag_a_h(xh, CH, row0, kc * 32, lane);

    const int nlo  = lane & 15;
    const int moff = (lane >= 16) ? 8 : 0;
#pragma unroll
    for (int nt = 0; nt < 8; ++nt) {
        const int n0 = nt * 16;
        const float bval = bias[n0 + nlo];
        v8f acc;
#pragma unroll
        for (int i = 0; i < 8; ++i) acc[i] = bval;
#pragma unroll
        for (int kc = 0; kc < 4; ++kc)
            acc = wmma16(afr[kc], frag_b_h(Bt, 136, n0, kc * 32, lane), acc);

        const int n   = n0 + nlo;
        const int d   = n & 31;
        const int g   = gbase + (n >> 5);   // group index within batch
#pragma unroll
        for (int i = 0; i < 8; ++i) {
            const int row = row0 + moff + i;
            const int b = row >> 9, s = row & 511;
            dst[(((long)(b * grp + g) * CS + s)) * CHD + d] = (_Float16)(acc[i] * scale);
        }
    }
}

// ---------------- Fused attention (scores + relations + softmax + P@V) ----------------
// grid 1024 = B*NH*(S/32), block 128 (4 waves). LDS ~99KB (< 320KB/WGP).

static constexpr int SD  = 516;  // scores row stride (f32), 4 mod 64 banks -> conflict-free tile stores
static constexpr int VTS = 520;  // vt row stride (f16), 1040B = 65*16B (b128-aligned)

__global__ __launch_bounds__(128) void k_attn(
        const _Float16* __restrict__ qh, const _Float16* __restrict__ kh,
        const _Float16* __restrict__ vh, const _Float16* __restrict__ mh,
        const float* __restrict__ mask, _Float16* __restrict__ cth) {
    __shared__ float    sc[32 * SD];
    __shared__ _Float16 vt[32 * VTS];

    const int tid  = threadIdx.x;
    const int lane = tid & 31;
    const int wv   = tid >> 5;
    const int bx   = blockIdx.x;
    const int qb = bx & 15, nh = (bx >> 4) & 3, b = bx >> 6;
    const int q0 = qb * 32;

    const long base = ((long)(b * CNH + nh)) * CS * CHD;
    const _Float16* Q  = qh + base;
    const _Float16* K  = kh + base;
    const _Float16* V  = vh + base;
    const long RSTR = (long)CNH * CS * CHD;                       // stride between relations
    const _Float16* M0 = mh + ((long)(b * CR) * CNH + nh) * CS * CHD;
    const float* Mk = mask + (long)b * CS * CS;

    // Stage V transposed: vt[d][t] (independent of score phase)
    for (int idx = tid; idx < CHD * CS; idx += 128) {
        int t = idx >> 5, d = idx & 31;
        vt[d * VTS + t] = V[(long)t * CHD + d];
    }

    // ---- Score phase: wave -> rowtile (wv&1), t-half (wv>>1); 16 tiles each ----
    const int rt = wv & 1;
    const int tg = wv >> 1;
    v16h aq = frag_a_h(Q, CHD, q0 + rt * 16, 0, lane);
    v16h am[4];
#pragma unroll
    for (int r = 0; r < 4; ++r) am[r] = frag_a_h(M0 + r * RSTR, CHD, q0 + rt * 16, 0, lane);

    const int nlo  = lane & 15;
    const int moff = (lane >= 16) ? 8 : 0;
    for (int it = 0; it < 16; ++it) {
        const int t0 = (tg * 16 + it) * 16;
        const int n = t0 + nlo;
        v8f c;
#pragma unroll
        for (int i = 0; i < 8; ++i) c[i] = 0.0f;   // mask folded into softmax pass
        c = wmma16(aq, frag_b_h(K, CHD, t0, 0, lane), c);
#pragma unroll
        for (int r = 0; r < 4; ++r)
            c = wmma16(am[r], frag_b_h(M0 + r * RSTR, CHD, t0, 0, lane), c);
#pragma unroll
        for (int i = 0; i < 8; ++i) sc[(rt * 16 + moff + i) * SD + n] = c[i];
    }
    __syncthreads();

    // ---- Mask add + softmax: 4 threads per row (same wave), shfl reductions ----
    {
        const int row = tid >> 2;        // 0..31
        const int sub = tid & 3;         // 128-col chunk
        float* rp = sc + row * SD + sub * 128;                 // 16B aligned (516*4 = 129*16)
        const float* mrow = Mk + (long)(q0 + row) * CS + sub * 128;  // contiguous, 16B aligned
        v4f* rp4 = (v4f*)rp;
        const v4f* mp4 = (const v4f*)mrow;
        float mx = -3.0e38f;
        for (int c = 0; c < 32; ++c) {
            v4f v = rp4[c] + mp4[c];
            rp4[c] = v;
            mx = fmaxf(mx, fmaxf(fmaxf(v[0], v[1]), fmaxf(v[2], v[3])));
        }
        mx = fmaxf(mx, __shfl_xor(mx, 1, 32));
        mx = fmaxf(mx, __shfl_xor(mx, 2, 32));
        float sum = 0.f;
        for (int c = 0; c < 32; ++c) {
            v4f v = rp4[c];
            v4f e;
#pragma unroll
            for (int j = 0; j < 4; ++j) e[j] = __expf(v[j] - mx);
            rp4[c] = e;
            sum += e[0] + e[1] + e[2] + e[3];
        }
        sum += __shfl_xor(sum, 1, 32);
        sum += __shfl_xor(sum, 2, 32);
        const float inv = 1.0f / sum;
        for (int c = 0; c < 32; ++c) rp4[c] *= inv;
    }
    __syncthreads();

    // ---- P@V: wave -> (rowtile, dtile); K over full S in 16 chunks of 32 ----
    const int prt = wv & 1;
    const int dt  = wv >> 1;
    v8f acc;
#pragma unroll
    for (int i = 0; i < 8; ++i) acc[i] = 0.f;
    for (int kt = 0; kt < 16; ++kt) {
        v16h ap = frag_a_f32(sc, SD, prt * 16, kt * 32, lane);
        v16h bvv = frag_b_h(vt, VTS, dt * 16, kt * 32, lane);
        acc = wmma16(ap, bvv, acc);
    }
#pragma unroll
    for (int i = 0; i < 8; ++i) {
        const int row = q0 + prt * 16 + moff + i;
        cth[((long)(b * CS + row)) * CH + nh * CHD + dt * 16 + nlo] = (_Float16)acc[i];
    }
}

// ---------------- Output projection + residual + LayerNorm ----------------
// grid 128, block 128 (4 waves). LDS aliased: Wd^T (34816B) then hidden (32768B).

__global__ __launch_bounds__(128) void k_out(
        const _Float16* __restrict__ cth, const float* __restrict__ Wd,
        const float* __restrict__ bd, const float* __restrict__ x,
        const float* __restrict__ lnw, const float* __restrict__ lnb,
        float* __restrict__ out) {
    __shared__ __align__(16) char smem[34816];
    _Float16* Bt  = (_Float16*)smem;     // [128][136]
    float*    hid = (float*)smem;        // [64][128], reused after GEMM

    const int tid  = threadIdx.x;
    const int lane = tid & 31;
    const int wv   = tid >> 5;

    for (int idx = tid; idx < 16384; idx += 128) { int k = idx >> 7, n = idx & 127; Bt[n * 136 + k] = (_Float16)Wd[idx]; }
    __syncthreads();

    const int row0g = blockIdx.x * 64;
    const int row0  = row0g + wv * 16;
    v16h afr[4];
#pragma unroll
    for (int kc = 0; kc < 4; ++kc) afr[kc] = frag_a_h(cth, CH, row0, kc * 32, lane);

    v8f accs[8];
#pragma unroll
    for (int nt = 0; nt < 8; ++nt) {
        const float bval = bd[nt * 16 + (lane & 15)];
        v8f a;
#pragma unroll
        for (int i = 0; i < 8; ++i) a[i] = bval;
#pragma unroll
        for (int kc = 0; kc < 4; ++kc)
            a = wmma16(afr[kc], frag_b_h(Bt, 136, nt * 16, kc * 32, lane), a);
        accs[nt] = a;
    }
    __syncthreads();   // done with Bt; reuse as hid

    const int nlo  = lane & 15;
    const int moff = (lane >= 16) ? 8 : 0;
#pragma unroll
    for (int nt = 0; nt < 8; ++nt)
#pragma unroll
        for (int i = 0; i < 8; ++i)
            hid[(wv * 16 + moff + i) * CH + nt * 16 + nlo] = accs[nt][i];
    __syncthreads();

    // LayerNorm (TF-style biased variance, eps 1e-12), residual add. One row per thread.
    if (tid < 64) {
        const long row = row0g + tid;
        float* hr = hid + tid * CH;
        const float* xr = x + row * CH;
        float sum = 0.f;
        for (int c = 0; c < CH; ++c) { float v = hr[c] + xr[c]; hr[c] = v; sum += v; }
        const float mean = sum * (1.0f / CH);
        float var = 0.f;
        for (int c = 0; c < CH; ++c) { float d = hr[c] - mean; var += d * d; }
        var *= (1.0f / CH);
        const float inv = rsqrtf(var + 1e-12f);
        float* orow = out + row * CH;
        for (int c = 0; c < CH; ++c) orow[c] = lnw[c] * (hr[c] - mean) * inv + lnb[c];
    }
}

// ---------------- Launch ----------------

extern "C" void kernel_launch(void* const* d_in, const int* in_sizes, int n_in,
                              void* d_out, int out_size, void* d_ws, size_t ws_size,
                              hipStream_t stream) {
    (void)in_sizes; (void)n_in; (void)out_size; (void)ws_size;
    const float* x    = (const float*)d_in[0];
    const float* mask = (const float*)d_in[1];
    // d_in[2] input_relationships_masks: unused by the reference
    const float* Wq = (const float*)d_in[3];  const float* bq = (const float*)d_in[4];
    const float* Wk = (const float*)d_in[5];  const float* bk = (const float*)d_in[6];
    const float* Wv = (const float*)d_in[7];  const float* bv = (const float*)d_in[8];
    const float* rel = (const float*)d_in[9]; const float* rw = (const float*)d_in[10];
    const float* Wd = (const float*)d_in[11]; const float* bd = (const float*)d_in[12];
    const float* lnw = (const float*)d_in[13]; const float* lnb = (const float*)d_in[14];
    float* out = (float*)d_out;

    char* ws = (char*)d_ws;
    float*    wp  = (float*)(ws + 0);           // 4 f32 softmax(rel_weights)
    float*    zb  = (float*)(ws + 256);         // 128 f32 zeros (bias for m_r)
    _Float16* xh  = (_Float16*)(ws + 1024);     // 2MB
    float*    sym = (float*)(ws + 2098176);     // 256KB
    _Float16* qh  = (_Float16*)(ws + 2360320);  // 2MB [B,NH,S,HD]
    _Float16* kh  = (_Float16*)(ws + 4457472);  // 2MB
    _Float16* vh  = (_Float16*)(ws + 6554624);  // 2MB
    _Float16* mh  = (_Float16*)(ws + 8651776);  // 8MB [B,R,NH,S,HD]
    _Float16* cth = (_Float16*)(ws + 17040384); // 2MB [B,S,H]

    k_wprob<<<1, 32, 0, stream>>>(rw, wp, zb);
    k_xh<<<4096, 256, 0, stream>>>(x, xh);
    k_sym<<<256, 256, 0, stream>>>(rel, sym);
    k_proj<<<dim3(128, 1, 7), 128, 0, stream>>>(xh, Wq, bq, Wk, bk, Wv, bv, sym, wp, zb,
                                                qh, kh, vh, mh);
    k_attn<<<1024, 128, 0, stream>>>(qh, kh, vh, mh, mask, cth);
    k_out<<<128, 128, 0, stream>>>(cth, Wd, bd, x, lnw, lnb, out);
}
